// GRUEncoder_47923245089058
// MI455X (gfx1250) — matrix operand order repaired and checked by reading, not compiled, not run on hardware
//
#include <hip/hip_runtime.h>

// Problem constants
#define BB   64
#define TT   4096
#define DD   128
#define HH   256
#define GG   768            // 3*H
#define CC   128
#define TC   256            // timesteps per chunk
#define NCHUNK (TT / TC)    // 16
#define NCT  (GG / 16)      // 48 column tiles of the gate dimension
#define NKD  (DD / 32)      // 4  k-steps (bf16 k=32) for input GEMM
#define NKH  (HH / 32)      // 8  k-steps for recurrent GEMM

// Workspace layout (bytes)
#define IG_BYTES   (TC * BB * GG * 2)            // 25,165,824 : igates chunk (bf16, C-frag order)
#define WIH_DW     (NCT * NKD * 8 * 32)          // 49,152 dwords
#define WHH_DW     (NCT * NKH * 8 * 32)          // 98,304 dwords
#define WIH_OFF    (IG_BYTES)
#define WHH_OFF    (WIH_OFF + WIH_DW * 4)
#define HST_OFF    (WHH_OFF + WHH_DW * 4)        // 16,384 floats (h state, frag order)
#define SMEM_BYTES ((32 * NKH * 8 * 32 + NKH * 8 * 32) * 4)  // 262144 + 8192 = 270,336

typedef __attribute__((ext_vector_type(16))) __bf16         bf16x16;
typedef __attribute__((ext_vector_type(8)))  float          f32x8;
typedef __attribute__((ext_vector_type(8)))  unsigned short u16x8;

__device__ __forceinline__ unsigned short f2bfu(float f) {
  return __builtin_bit_cast(unsigned short, (__bf16)f);   // RNE convert
}
__device__ __forceinline__ unsigned pack_bf2(float lo, float hi) {
  return (unsigned)f2bfu(lo) | ((unsigned)f2bfu(hi) << 16);
}
__device__ __forceinline__ float bfu2f(unsigned short u) {
  unsigned v = ((unsigned)u) << 16;
  return __builtin_bit_cast(float, v);
}
__device__ __forceinline__ float fsigmoid(float x) { return 1.0f / (1.0f + __expf(-x)); }
__device__ __forceinline__ float ftanh_(float x) {
  float e = __expf(-2.0f * x);
  return (1.0f - e) / (1.0f + e);
}

// ---------------------------------------------------------------------------
// Fragment layouts (lane-major: each lane's 8 dwords are contiguous = 32B):
//   B frag dword index = ((coltile*NK + k)*32 + lane)*8 + v
//     bf16 pair in dword v of lane:  N = lane%16,
//     K = k*32 + (v<4?0:16) + (lane>=16?8:0) + (v&3)*2 + {0,1}
//   A frag (h in LDS):  idx = (chunk*32 + lane)*8 + v  ; per ISA A table:
//     M = lane%16, K = chunk*32 + (v<4?0:16) + (lane>=16?8:0) + (v&3)*2 + half
//   C frag (igates):  u16x8 index = (rowtile*NCT + ct)*32 + lane
//     element v: (M = v + 8*(lane>=16), N = lane%16)
// ---------------------------------------------------------------------------

// Kernel 1: pre-swizzle w_ih / w_hh into B-fragment order (bf16 pairs).
__global__ void gru_prep(const float* __restrict__ w_ih, const float* __restrict__ w_hh,
                         unsigned* __restrict__ wih_swz, unsigned* __restrict__ whh_swz) {
  int idx = blockIdx.x * blockDim.x + threadIdx.x;
  if (idx < WIH_DW) {
    int v = idx & 7, lane = (idx >> 3) & 31, k = (idx >> 8) & 3, ct = idx >> 10;
    int g  = ct * 16 + (lane & 15);
    int kg = k * 32 + ((lane >= 16) ? 8 : 0) + ((v < 4) ? 0 : 16) + (v & 3) * 2;
    wih_swz[idx] = pack_bf2(w_ih[g * DD + kg], w_ih[g * DD + kg + 1]);
  } else if (idx < WIH_DW + WHH_DW) {
    int j = idx - WIH_DW;
    int v = j & 7, lane = (j >> 3) & 31, k = (j >> 8) & 7, ct = j >> 11;
    int g  = ct * 16 + (lane & 15);
    int kg = k * 32 + ((lane >= 16) ? 8 : 0) + ((v < 4) ? 0 : 16) + (v & 3) * 2;
    whh_swz[j] = pack_bf2(w_hh[g * HH + kg], w_hh[g * HH + kg + 1]);
  }
}

// ---------------------------------------------------------------------------
// Kernel 2: igates chunk GEMM.  Row order [tl][b]: one 16-row tile = 16 batch
// rows at one timestep (what the scan consumes).  Output bf16, C-frag order,
// 16B per lane per tile (single b128 store).
// ---------------------------------------------------------------------------
__global__ void __launch_bounds__(128)
gru_igates(const float* __restrict__ x, const unsigned* __restrict__ wih_swz,
           const float* __restrict__ bias, unsigned short* __restrict__ ig, int t0) {
  int lane = threadIdx.x & 31;
  int wid  = threadIdx.x >> 5;
  int rt   = blockIdx.x * 4 + wid;           // row tile 0..1023
  int tl   = rt >> 2;                         // timestep within chunk
  int bt   = rt & 3;                          // batch tile
  int m    = lane & 15;
  int b    = bt * 16 + m;
  long xrow = ((long)b * TT + (t0 + tl)) * DD;

  const bf16x16* bfr = (const bf16x16*)wih_swz;

  // A fragments: K spans of 8 consecutive floats -> float4 pairs, cvt to bf16
  bf16x16 afrag[NKD];
#pragma unroll
  for (int c = 0; c < NKD; ++c) {
    bf16x16 a;
#pragma unroll
    for (int hg = 0; hg < 2; ++hg) {
      int kb = c * 32 + hg * 16 + ((lane >= 16) ? 8 : 0);
      float4 x0 = *(const float4*)(x + xrow + kb);
      float4 x1 = *(const float4*)(x + xrow + kb + 4);
      a[8 * hg + 0] = (__bf16)x0.x;  a[8 * hg + 1] = (__bf16)x0.y;
      a[8 * hg + 2] = (__bf16)x0.z;  a[8 * hg + 3] = (__bf16)x0.w;
      a[8 * hg + 4] = (__bf16)x1.x;  a[8 * hg + 5] = (__bf16)x1.y;
      a[8 * hg + 6] = (__bf16)x1.z;  a[8 * hg + 7] = (__bf16)x1.w;
    }
    afrag[c] = a;
  }

  for (int ct = 0; ct < NCT; ++ct) {
    float bv = bias[ct * 16 + m];
    f32x8 acc = {bv, bv, bv, bv, bv, bv, bv, bv};
#pragma unroll
    for (int k = 0; k < NKD; ++k) {
      bf16x16 bfrag = bfr[(ct * NKD + k) * 32 + lane];     // 2x b128, coalesced
      acc = __builtin_amdgcn_wmma_f32_16x16x32_bf16(false, afrag[k], false, bfrag,
                                                    (short)0, acc, false, false);
    }
    u16x8 o;
#pragma unroll
    for (int v = 0; v < 8; ++v) o[v] = f2bfu(acc[v]);
    ((u16x8*)ig)[((long)rt * NCT + ct) * 32 + lane] = o;    // 1x b128 store
  }
}

// ---------------------------------------------------------------------------
// Kernel 3: persistent GRU scan over one chunk of TC steps.
// 4 WGs x 256 threads; WG wg owns batch rows [wg*16, wg*16+16) -> zero
// inter-WG sync.  r,z weights (256 KB) in LDS; n-gate weights held in
// REGISTERS (loop-invariant, 128 VGPRs/lane) -> no per-step global weight
// traffic.  h carried fp32 in registers, mirrored bf16 in LDS (A-frag order).
// Wave w owns H-columns [w*32, w*32+32): gate tiles ct = q*16 + w*2 + s.
// ---------------------------------------------------------------------------
__global__ void __launch_bounds__(256, 1)
gru_scan(const unsigned short* __restrict__ ig, const unsigned* __restrict__ whh_swz,
         const float* __restrict__ bn, float* __restrict__ hstate, int first) {
  extern __shared__ unsigned smem[];
  unsigned* wrz  = smem;                        // 65,536 dwords: gates r,z (ct 0..31)
  unsigned* hbuf = smem + 32 * NKH * 8 * 32;    // 2,048 dwords: h, A-frag order

  const int tid = threadIdx.x, lane = tid & 31, w = tid >> 5, wg = blockIdx.x;

  // Stage r,z weights into LDS (coalesced; already fragment-ordered)
  for (int i = tid; i < 32 * NKH * 8 * 32; i += 256) wrz[i] = whh_swz[i];

  // Loop-invariant n-gate B fragments -> registers (2 tiles x 8 k)
  bf16x16 nB[2][NKH];
#pragma unroll
  for (int s = 0; s < 2; ++s)
#pragma unroll
    for (int k = 0; k < NKH; ++k)
      nB[s][k] = ((const bf16x16*)whh_swz)[(((32 + w * 2 + s) * NKH + k) * 32) + lane];

  // Carried hidden state (fp32): element (m = v + 8*(lane>=16), j = w*32+s*16+(lane&15))
  float hreg[2][8];
  if (first) {
#pragma unroll
    for (int s = 0; s < 2; ++s)
#pragma unroll
      for (int v = 0; v < 8; ++v) hreg[s][v] = 0.0f;
  } else {
#pragma unroll
    for (int s = 0; s < 2; ++s)
#pragma unroll
      for (int v = 0; v < 8; ++v)
        hreg[s][v] = hstate[((((wg * 8 + w) * 2 + s) * 8) + v) * 32 + lane];
  }
  float bnv[2];
#pragma unroll
  for (int s = 0; s < 2; ++s) bnv[s] = bn[w * 32 + s * 16 + (lane & 15)];

  // Scatter h (bf16) into hbuf, A-fragment order.  A-chunk for this wave = w.
  const int nl    = lane & 15;
  const int half  = nl & 1;
  const int hiA   = (nl >= 8) ? 1 : 0;
  const int mbase = (lane >= 16) ? 8 : 0;
  auto store_h = [&](int s, int v, float val) {
    int m     = v + mbase;
    int vA    = s * 4 + ((nl & 7) >> 1);
    int laneA = m + 16 * hiA;
    unsigned dw = (unsigned)((w * 32 + laneA) * 8 + vA);
    ((unsigned short*)hbuf)[dw * 2 + half] = f2bfu(val);
  };
#pragma unroll
  for (int s = 0; s < 2; ++s)
#pragma unroll
    for (int v = 0; v < 8; ++v) store_h(s, v, hreg[s][v]);
  __syncthreads();

  const bf16x16* hA  = (const bf16x16*)hbuf;
  const bf16x16* rzB = (const bf16x16*)wrz;
  const u16x8*   igf8 = (const u16x8*)ig;

  for (int tl = 0; tl < TC; ++tl) {
    const int rt = tl * 4 + wg;

    // Prefetch next step's igate fragments (global_prefetch_b8)
    if (tl + 1 < TC) {
      const unsigned short* np = ig + ((long)(rt + 4) * NCT) * 256;
      __builtin_prefetch(np + tid * 48, 0, 0);
    }

    // This step's igate fragments: 6 coalesced b128 loads
    u16x8 igv[3][2];
#pragma unroll
    for (int q = 0; q < 3; ++q)
#pragma unroll
      for (int s = 0; s < 2; ++s)
        igv[q][s] = igf8[((long)rt * NCT + (q * 16 + w * 2 + s)) * 32 + lane];

    // Recurrent GEMM: k outer, 6 resident accumulators, A loaded per-k
    f32x8 acc[3][2];
#pragma unroll
    for (int q = 0; q < 3; ++q)
#pragma unroll
      for (int s = 0; s < 2; ++s) acc[q][s] = (f32x8){};

#pragma unroll
    for (int k = 0; k < NKH; ++k) {
      bf16x16 ak = hA[k * 32 + lane];                       // 2x ds_load_b128
#pragma unroll
      for (int q = 0; q < 3; ++q) {
#pragma unroll
        for (int s = 0; s < 2; ++s) {
          bf16x16 bfrag = (q < 2)
              ? rzB[((q * 16 + w * 2 + s) * NKH + k) * 32 + lane]  // LDS
              : nB[s][k];                                          // registers
          acc[q][s] = __builtin_amdgcn_wmma_f32_16x16x32_bf16(false, ak, false, bfrag,
                                                              (short)0, acc[q][s],
                                                              false, false);
        }
      }
    }
    __syncthreads();  // all A reads of h(t) done; hbuf may be overwritten

    // Gates + state update (fp32), h(t+1) back into hbuf as bf16
#pragma unroll
    for (int s = 0; s < 2; ++s) {
#pragma unroll
      for (int v = 0; v < 8; ++v) {
        float r = fsigmoid(bfu2f(igv[0][s][v]) + acc[0][s][v]);
        float z = fsigmoid(bfu2f(igv[1][s][v]) + acc[1][s][v]);
        float n = ftanh_(bfu2f(igv[2][s][v]) + r * (acc[2][s][v] + bnv[s]));
        float h = n + z * (hreg[s][v] - n);
        hreg[s][v] = h;
        store_h(s, v, h);
      }
    }
    __syncthreads();  // h(t+1) fully published before next A load
  }

  // Persist h for the next chunk / projection (fragment order, coalesced)
#pragma unroll
  for (int s = 0; s < 2; ++s)
#pragma unroll
    for (int v = 0; v < 8; ++v)
      hstate[((((wg * 8 + w) * 2 + s) * 8) + v) * 32 + lane] = hreg[s][v];
}

// ---------------------------------------------------------------------------
// Kernel 4: final projection out = h @ w_projT + b_proj  (fp32, tiny)
// ---------------------------------------------------------------------------
__global__ void gru_proj(const float* __restrict__ hstate, const float* __restrict__ w_proj,
                         const float* __restrict__ b_proj, float* __restrict__ out) {
  int idx = blockIdx.x * blockDim.x + threadIdx.x;
  if (idx >= BB * CC) return;
  int b = idx / CC, c = idx % CC;
  int wg = b >> 4, m = b & 15;
  float sum = b_proj[c];
  for (int j = 0; j < HH; ++j) {
    int w = j >> 5, s = (j >> 4) & 1, nl = j & 15;
    int lane = nl + ((m >= 8) ? 16 : 0);
    float h = hstate[((((wg * 8 + w) * 2 + s) * 8) + (m & 7)) * 32 + lane];
    sum += h * w_proj[c * HH + j];
  }
  out[idx] = sum;
}

// ---------------------------------------------------------------------------
extern "C" void kernel_launch(void* const* d_in, const int* in_sizes, int n_in,
                              void* d_out, int out_size, void* d_ws, size_t ws_size,
                              hipStream_t stream) {
  (void)in_sizes; (void)n_in; (void)out_size; (void)ws_size;
  const float* x      = (const float*)d_in[0];
  const float* w_ih   = (const float*)d_in[1];
  const float* w_hh   = (const float*)d_in[2];
  const float* bias   = (const float*)d_in[3];
  const float* bn     = (const float*)d_in[4];
  const float* w_proj = (const float*)d_in[5];
  const float* b_proj = (const float*)d_in[6];

  char* ws = (char*)d_ws;
  unsigned short* ig      = (unsigned short*)(ws);
  unsigned*       wih_swz = (unsigned*)(ws + WIH_OFF);
  unsigned*       whh_swz = (unsigned*)(ws + WHH_OFF);
  float*          hstate  = (float*)(ws + HST_OFF);

  // 270 KB dynamic LDS (of the 320 KB/WGP) for the persistent scan kernel
  (void)hipFuncSetAttribute((const void*)gru_scan,
                            hipFuncAttributeMaxDynamicSharedMemorySize, SMEM_BYTES);

  gru_prep<<<(WIH_DW + WHH_DW + 255) / 256, 256, 0, stream>>>(w_ih, w_hh, wih_swz, whh_swz);
  for (int c = 0; c < NCHUNK; ++c) {
    gru_igates<<<256, 128, 0, stream>>>(x, wih_swz, bias, ig, c * TC);
    gru_scan<<<4, 256, SMEM_BYTES, stream>>>(ig, whh_swz, bn, hstate, c == 0 ? 1 : 0);
  }
  gru_proj<<<(BB * CC + 255) / 256, 256, 0, stream>>>(hstate, w_proj, b_proj, (float*)d_out);
}
